// CapsuleNetwork_12927851561038
// MI455X (gfx1250) — matrix operand order repaired and checked by reading, not compiled, not run on hardware
//
#include <hip/hip_runtime.h>
#include <hip/hip_bf16.h>
#include <math.h>

typedef __attribute__((ext_vector_type(16))) _Float16       v16h;
typedef __attribute__((ext_vector_type(8)))  float          v8f;
typedef __attribute__((ext_vector_type(2)))  float          v2f;
typedef __attribute__((ext_vector_type(4)))  unsigned short us4;
typedef int v4i __attribute__((vector_size(16)));   // matches builtin param type

union FragH { v16h h; uint4 q2[2]; us4 q[4]; };

// ---- optional CDNA5 async global->LDS path (guarded; fallback compiles) ----
#if defined(__has_builtin)
#if __has_builtin(__builtin_amdgcn_global_load_async_to_lds_b128) && \
    __has_builtin(__builtin_amdgcn_s_wait_asynccnt)
#define HAVE_ASYNC_LDS 1
#endif
#endif

#ifdef HAVE_ASYNC_LDS
__device__ __forceinline__ void async_ld16(void* lds, const void* g) {
  __builtin_amdgcn_global_load_async_to_lds_b128(
      (__attribute__((address_space(1))) v4i*)(v4i*)(void*)g,
      (__attribute__((address_space(3))) v4i*)(v4i*)lds, 0, 0);
}
#endif

// ---------------------------------------------------------------------------
// Kernel 1: tiled transpose + f32->f16: conv2 weights [K=20736][N=256]
//           -> w2t [N=256][K=20736] so GEMM B-tiles stage as contiguous b128.
// ---------------------------------------------------------------------------
__global__ __launch_bounds__(256) void transpose_w2_f16(
    const float* __restrict__ in, _Float16* __restrict__ out) {
  __shared__ float tile[32][33];
  const int k0 = blockIdx.x * 32;
  const int n0 = blockIdx.y * 32;
  const int tx = threadIdx.x;            // 0..31
  const int ty = threadIdx.y;            // 0..7
  for (int r = ty; r < 32; r += 8)
    tile[r][tx] = in[(long)(k0 + r) * 256 + n0 + tx];
  __syncthreads();
  for (int r = ty; r < 32; r += 8)
    out[(long)(n0 + r) * 20736 + k0 + tx] = (_Float16)tile[tx][r];
}

// ---------------------------------------------------------------------------
// Kernel 2: conv1 (9x9x1->256, stride 1) + ReLU, f16 output.
// One block per (b, oy); each thread keeps 20 ox accumulators in registers.
// ---------------------------------------------------------------------------
__global__ __launch_bounds__(256) void conv1_relu_f16(
    const float* __restrict__ x, const float* __restrict__ k1,
    const float* __restrict__ b1, _Float16* __restrict__ x1) {
  const int bid = blockIdx.x;            // b*20 + oy
  const int b = bid / 20, oy = bid % 20;
  const int t = threadIdx.x;             // channel
  __shared__ float patch[9 * 28];
  if (t < 252) {
    int r = t / 28, col = t % 28;
    patch[t] = x[b * 784 + (oy + r) * 28 + col];
  }
  __syncthreads();
  float acc[20];
#pragma unroll
  for (int i = 0; i < 20; ++i) acc[i] = 0.f;
  for (int ky = 0; ky < 9; ++ky) {
#pragma unroll
    for (int kx = 0; kx < 9; ++kx) {
      float kv = k1[(ky * 9 + kx) * 256 + t];
#pragma unroll
      for (int ox = 0; ox < 20; ++ox)
        acc[ox] = fmaf(patch[ky * 28 + ox + kx], kv, acc[ox]);
    }
  }
  float bv = b1[t];
#pragma unroll
  for (int ox = 0; ox < 20; ++ox) {
    float v = acc[ox] + bv;
    v = v > 0.f ? v : 0.f;
    x1[((b * 20 + oy) * 20 + ox) * 256 + t] = (_Float16)v;
  }
}

// ---------------------------------------------------------------------------
// Kernel 3: conv2 as implicit GEMM, f16 WMMA, f32 accumulate.
// M = 9216 (b,oy,ox), N = 256 (co), K = 20736 (kh,kw,ci).
// Block tile 128x64, Ktile 32, 8 waves (4x2), each wave 32x32 (2x2 tiles).
// Double-buffered LDS; rows padded to 56 f16 (112B: b128-aligned AND
// 28-dword stride -> 16 distinct banks for fragment reads).
// ---------------------------------------------------------------------------
#define LDPAD 56
#define NITER 648

__global__ __launch_bounds__(256) void conv2_wmma_f16(
    const _Float16* __restrict__ x1, const _Float16* __restrict__ w2t,
    const float* __restrict__ b2, float* __restrict__ out) {
  __shared__ _Float16 Alds[2][128 * LDPAD];
  __shared__ _Float16 Blds[2][64 * LDPAD];

  const int t    = threadIdx.x;
  const int lane = t & 31;
  const int wv   = t >> 5;
  const int wm   = wv >> 1;              // 0..3  (M quadrant)
  const int wn   = wv & 1;               // 0..1  (N half)
  const int hi   = lane >> 4;
  const int l15  = lane & 15;
  const int M0   = blockIdx.y * 128;
  const int N0   = blockIdx.x * 64;

  // A staging: 2 threads per row, 32B (2 x b128) each
  const int arow  = t >> 1;
  const int ahalf = t & 1;
  const int gm = M0 + arow;
  const int ab = gm / 36;
  const int rr = gm % 36;
  const long abase = (long)((ab * 20 + 2 * (rr / 6)) * 20 + 2 * (rr % 6)) * 256;
  const int aoff = arow * LDPAD + ahalf * 16;

  // B staging: 4 threads per row, 16B (1 x b128) each
  const int brow = t >> 2;               // 0..63
  const int bq   = t & 3;                // 0..3
  const long bbase = (long)(N0 + brow) * 20736 + bq * 8;
  const int boff = brow * LDPAD + bq * 8;

  v8f zero = {};
  v8f acc[2][2];
  acc[0][0] = zero; acc[0][1] = zero; acc[1][0] = zero; acc[1][1] = zero;

  // address of this thread's A chunk for K-block starting at k0
  auto aSrc = [&](int k0) -> const _Float16* {
    const int p = k0 >> 8;               // kh*9 + kw
    return x1 + abase + ((p / 9) * 20 + (p % 9)) * 256 + (k0 & 255) + ahalf * 16;
  };

  // ---- preload tile 0 into buffer 0 ----
  {
    const _Float16* as = aSrc(0);
    const _Float16* bs = w2t + bbase;
#ifdef HAVE_ASYNC_LDS
    async_ld16(&Alds[0][aoff], as);
    async_ld16(&Alds[0][aoff + 8], as + 8);
    async_ld16(&Blds[0][boff], bs);
    __builtin_amdgcn_s_wait_asynccnt(0);
#else
    uint4 ra0 = ((const uint4*)as)[0];
    uint4 ra1 = ((const uint4*)as)[1];
    uint4 rb0 = ((const uint4*)bs)[0];
    uint4* ad = (uint4*)&Alds[0][aoff];
    ad[0] = ra0; ad[1] = ra1;
    *(uint4*)&Blds[0][boff] = rb0;
#endif
  }
  __syncthreads();

  for (int it = 0; it < NITER; ++it) {
    const int cur = it & 1, nxt = cur ^ 1;
    const bool haveNext = (it + 1 < NITER);
    const int kn = (it + 1) * 32;

#ifdef HAVE_ASYNC_LDS
    if (haveNext) {
      const _Float16* as = aSrc(kn);
      const _Float16* bs = w2t + bbase + kn;
      async_ld16(&Alds[nxt][aoff], as);
      async_ld16(&Alds[nxt][aoff + 8], as + 8);
      async_ld16(&Blds[nxt][boff], bs);
    }
#else
    uint4 ra0, ra1, rb0;
    if (haveNext) {
      const _Float16* as = aSrc(kn);
      const _Float16* bs = w2t + bbase + kn;
      ra0 = ((const uint4*)as)[0];
      ra1 = ((const uint4*)as)[1];
      rb0 = ((const uint4*)bs)[0];
    }
#endif

    // ---- fragments from current buffer ----
    FragH a[2], bf[2];
#pragma unroll
    for (int mt = 0; mt < 2; ++mt) {
      const _Float16* ap = &Alds[cur][(wm * 32 + mt * 16 + l15) * LDPAD];
      a[mt].q2[0] = *(const uint4*)(ap + 8 * hi);        // K = 8*hi .. +7
      a[mt].q2[1] = *(const uint4*)(ap + 16 + 8 * hi);   // K = 16+8*hi .. +7
    }
#pragma unroll
    for (int nt = 0; nt < 2; ++nt) {
      const _Float16* bp =
          &Blds[cur][(wn * 32 + nt * 16 + l15) * LDPAD + 8 * hi];
      bf[nt].q2[0] = *(const uint4*)(bp);                // K = 8*hi .. +7
      bf[nt].q2[1] = *(const uint4*)(bp + 16);           // K = 16+8*hi .. +7
    }
#pragma unroll
    for (int mt = 0; mt < 2; ++mt)
#pragma unroll
      for (int nt = 0; nt < 2; ++nt)
        acc[mt][nt] = __builtin_amdgcn_wmma_f32_16x16x32_f16(
            false, a[mt].h, false, bf[nt].h, (short)0, acc[mt][nt],
            false, false);

#ifdef HAVE_ASYNC_LDS
    if (haveNext) __builtin_amdgcn_s_wait_asynccnt(0);
#else
    if (haveNext) {
      uint4* ad = (uint4*)&Alds[nxt][aoff];
      ad[0] = ra0; ad[1] = ra1;
      *(uint4*)&Blds[nxt][boff] = rb0;
    }
#endif
    __syncthreads();
  }

  // ---- epilogue: bias add, store f32 (= u reshaped (B,1152,8)) ----
#pragma unroll
  for (int mt = 0; mt < 2; ++mt) {
    const int gmr = M0 + wm * 32 + mt * 16 + 8 * hi;
#pragma unroll
    for (int nt = 0; nt < 2; ++nt) {
      const int gn = N0 + wn * 32 + nt * 16 + l15;
      const float bias = b2[gn];
#pragma unroll
      for (int r = 0; r < 8; ++r)
        out[(long)(gmr + r) * 256 + gn] = acc[mt][nt][r] + bias;
    }
  }
}

// ---------------------------------------------------------------------------
// Kernel 4: u_hat[b,i,j,k] = sum_l w[i,j,k,l] * u[b,i,l]
// One block per capsule i; thread = batch element b.
// ---------------------------------------------------------------------------
__global__ __launch_bounds__(256) void uhat_kernel(
    const float* __restrict__ u, const float* __restrict__ w,
    float* __restrict__ uhat) {
  const int i = blockIdx.x;
  const int t = threadIdx.x;   // b
  __shared__ float wl[1280];   // w[i] : 160 x 8
  for (int j = t; j < 1280; j += 256) wl[j] = w[(long)i * 1280 + j];
  __syncthreads();
  float ur[8];
  const float* up = u + ((long)t * 1152 + i) * 8;
#pragma unroll
  for (int l = 0; l < 8; ++l) ur[l] = up[l];
  float* op = uhat + ((long)t * 1152 + i) * 160;
  for (int jk = 0; jk < 160; ++jk) {
    float acc = 0.f;
#pragma unroll
    for (int l = 0; l < 8; ++l) acc = fmaf(wl[jk * 8 + l], ur[l], acc);
    op[jk] = acc;
  }
}

// ---------------------------------------------------------------------------
// Kernel 5: fused dynamic routing (3 iters), one block per batch element.
// ---------------------------------------------------------------------------
__global__ __launch_bounds__(256) void routing_kernel(
    const float* __restrict__ uhat, const float* __restrict__ y,
    float* __restrict__ vout, float* __restrict__ vmasked) {
  const int b = blockIdx.x;
  const int t = threadIdx.x;
  __shared__ float blog[11520];   // logits [i][j]
  __shared__ float mz[2304];      // per-i (max, 1/Z)
  __shared__ float ssh[160];
  __shared__ float vsh[160];
  __shared__ float scl[10];
  for (int i = t; i < 11520; i += 256) blog[i] = 0.f;
  __syncthreads();
  const float* uh = uhat + (long)b * 184320;

  for (int iter = 0; iter < 3; ++iter) {
    for (int i = t; i < 1152; i += 256) {     // softmax stats over j
      float m = blog[i * 10];
      for (int j = 1; j < 10; ++j) m = fmaxf(m, blog[i * 10 + j]);
      float z = 0.f;
      for (int j = 0; j < 10; ++j) z += __expf(blog[i * 10 + j] - m);
      mz[i * 2] = m; mz[i * 2 + 1] = 1.f / z;
    }
    __syncthreads();
    if (t < 160) {                            // s[j,k] = sum_i c[i,j]*uhat
      const int j = t >> 4;
      float acc = 0.f;
      for (int i = 0; i < 1152; ++i) {
        float c = __expf(blog[i * 10 + j] - mz[2 * i]) * mz[2 * i + 1];
        acc = fmaf(c, uh[(long)i * 160 + t], acc);
      }
      ssh[t] = acc;
    }
    __syncthreads();
    if (t < 10) {                             // squash scale
      float n2 = 0.f;
      for (int k = 0; k < 16; ++k) { float s = ssh[t * 16 + k]; n2 = fmaf(s, s, n2); }
      scl[t] = n2 / ((1.f + n2) * (sqrtf(n2) + 1e-7f));
    }
    __syncthreads();
    if (t < 160) vsh[t] = ssh[t] * scl[t >> 4];
    __syncthreads();
    if (iter < 2) {                           // logit update
      for (int i = t; i < 1152; i += 256) {
        const float* ur = uh + (long)i * 160;
        for (int j = 0; j < 10; ++j) {
          float d = 0.f;
#pragma unroll
          for (int k = 0; k < 16; ++k) d = fmaf(ur[j * 16 + k], vsh[j * 16 + k], d);
          blog[i * 10 + j] += d;
        }
      }
      __syncthreads();
    }
  }
  if (t < 160) {
    float v = vsh[t];
    vout[(long)b * 160 + t] = v;
    vmasked[(long)b * 160 + t] = y[b * 10 + (t >> 4)] * v;
  }
}

// ---------------------------------------------------------------------------
// Kernel 6: decoder GEMM via native fp32 WMMA (16x16x4), fused bias+act.
// One wave per 16x16 C tile. act: 0=relu, 1=sigmoid.
// ---------------------------------------------------------------------------
__global__ __launch_bounds__(32) void gemm_wmma_f32(
    const float* __restrict__ A, const float* __restrict__ B,
    const float* __restrict__ bias, float* __restrict__ C,
    int M, int N, int K, int act) {
  const int lane = threadIdx.x & 31;
  const int hi = lane >> 4, l15 = lane & 15;
  const int n = blockIdx.x * 16 + l15;
  const int M0 = blockIdx.y * 16;
  const int m = M0 + l15;
  v8f acc = {};
  for (int k = 0; k < K; k += 4) {
    v2f av, bv;
    const float* ap = A + (long)m * K + k + 2 * hi;     // A[m][k+2hi .. +1]
    av[0] = ap[0]; av[1] = ap[1];
    const float* bp = B + (long)(k + 2 * hi) * N + n;   // B[k+2hi .. +1][n]
    bv[0] = bp[0]; bv[1] = bp[N];
    acc = __builtin_amdgcn_wmma_f32_16x16x4_f32(
        false, av, false, bv, (short)0, acc, false, false);
  }
  const float bb = bias[n];
#pragma unroll
  for (int r = 0; r < 8; ++r) {
    float v = acc[r] + bb;
    if (act == 0)      v = v > 0.f ? v : 0.f;
    else               v = 1.f / (1.f + __expf(-v));
    C[(long)(M0 + r + 8 * hi) * N + n] = v;
  }
}

// ---------------------------------------------------------------------------
extern "C" void kernel_launch(void* const* d_in, const int* in_sizes, int n_in,
                              void* d_out, int out_size, void* d_ws, size_t ws_size,
                              hipStream_t stream) {
  const float* x   = (const float*)d_in[0];   // (256,28,28,1)
  const float* y   = (const float*)d_in[1];   // (256,10)
  const float* k1  = (const float*)d_in[2];   // (9,9,1,256)
  const float* b1c = (const float*)d_in[3];   // (256,)
  const float* k2  = (const float*)d_in[4];   // (9,9,256,256)
  const float* b2c = (const float*)d_in[5];   // (256,)
  const float* w   = (const float*)d_in[6];   // (1152,10,16,8)
  const float* w1  = (const float*)d_in[7];   // (160,256)
  const float* bb1 = (const float*)d_in[8];
  const float* w2d = (const float*)d_in[9];   // (256,512)
  const float* bb2 = (const float*)d_in[10];
  const float* w3  = (const float*)d_in[11];  // (512,784)
  const float* bb3 = (const float*)d_in[12];
  float* out = (float*)d_out;                 // [v: 40960][recon: 200704]

  char* ws = (char*)d_ws;
  size_t off = 0;
  auto wsalloc = [&](size_t bytes) {
    void* p = ws + off;
    off += (bytes + 255) & ~(size_t)255;
    return p;
  };
  _Float16* x1h  = (_Float16*)wsalloc(26214400ull * 2);  // conv1 out f16
  _Float16* w2t  = (_Float16*)wsalloc(5308416ull * 2);   // conv2 weights f16, [N][K]
  float*    u    = (float*)wsalloc(2359296ull * 4);      // conv2 out / u
  float*    uhat = (float*)wsalloc(47185920ull * 4);     // u_hat
  float*    vm   = (float*)wsalloc(40960ull * 4);        // masked v
  float*    h1   = (float*)wsalloc(65536ull * 4);
  float*    h2   = (float*)wsalloc(131072ull * 4);

  transpose_w2_f16<<<dim3(648, 8), dim3(32, 8), 0, stream>>>(k2, w2t);
  conv1_relu_f16<<<5120, 256, 0, stream>>>(x, k1, b1c, x1h);
  conv2_wmma_f16<<<dim3(4, 72), 256, 0, stream>>>(x1h, w2t, b2c, u);
  uhat_kernel<<<1152, 256, 0, stream>>>(u, w, uhat);
  routing_kernel<<<256, 256, 0, stream>>>(uhat, y, out, vm);
  gemm_wmma_f32<<<dim3(256 / 16, 16), 32, 0, stream>>>(vm, w1, bb1, h1, 256, 256, 160, 0);
  gemm_wmma_f32<<<dim3(512 / 16, 16), 32, 0, stream>>>(h1, w2d, bb2, h2, 256, 512, 256, 0);
  gemm_wmma_f32<<<dim3(784 / 16, 16), 32, 0, stream>>>(h2, w3, bb3, out + 40960, 256, 784, 512, 1);
}
